// CausalSelfAttention_7164005450182
// MI455X (gfx1250) — matrix operand order repaired and checked by reading, not compiled, not run on hardware
//
#include <hip/hip_runtime.h>
#include <hip/hip_bf16.h>

typedef __attribute__((ext_vector_type(16))) _Float16 v16h;
typedef __attribute__((ext_vector_type(8)))  float    v8f;

union V16H { uint4 u[2]; v16h v; };
union H4   { _Float16 h[4]; uint2 u; };

__device__ __forceinline__ unsigned short f2h(float f) {
  union { _Float16 h; unsigned short u; } t; t.h = (_Float16)f; return t.u;
}

__device__ __forceinline__ v8f wmma_f16(v16h a, v16h b, v8f c) {
  // D = A(16x32 f16) * B(32x16 f16) + C(16x16 f32)
  return __builtin_amdgcn_wmma_f32_16x16x32_f16(false, a, false, b, (short)0, c,
                                                false, false);
}

// Async copy: 16B global -> LDS, tracked by ASYNCcnt (CDNA5 ISA).
// lds = wave-relative LDS byte offset (low 32 bits of generic shared pointer).
__device__ __forceinline__ unsigned lds_off(const void* p) {
  return (unsigned)(unsigned long long)p;
}
__device__ __forceinline__ void async_b128(unsigned lds, const void* gaddr) {
  asm volatile("global_load_async_to_lds_b128 %0, %1, off"
               :: "v"(lds), "v"(gaddr) : "memory");
}
__device__ __forceinline__ void wait_async0() {
  asm volatile("s_wait_asynccnt 0" ::: "memory");
}

// Problem constants (B=4, T=2048, C=1024, H=16, D=64)
#define DIM_B 4
#define DIM_T 2048
#define DIM_C 1024
#define DIM_H 16
#define DIM_D 64

// ---------------------------------------------------------------------------
// Prep A: elementwise f32 -> f16 (for x)
// ---------------------------------------------------------------------------
__global__ __launch_bounds__(256) void cvt_f16_kernel(
    const float* __restrict__ in, _Float16* __restrict__ out) {
  int idx = blockIdx.x * 256 + threadIdx.x;
  float4 v = *(const float4*)(in + (size_t)idx * 4);
  H4 pk;
  pk.h[0] = (_Float16)v.x; pk.h[1] = (_Float16)v.y;
  pk.h[2] = (_Float16)v.z; pk.h[3] = (_Float16)v.w;
  *(uint2*)(out + (size_t)idx * 4) = pk.u;
}

// ---------------------------------------------------------------------------
// Prep B: W f32 [1024][N] -> Wt f16 [N][1024] (LDS-tiled 32x32 transpose)
// ---------------------------------------------------------------------------
__global__ __launch_bounds__(256) void wtrans_kernel(
    const float* __restrict__ W, _Float16* __restrict__ Wt, int N) {
  __shared__ unsigned short Ts[32 * 36];  // padded rows to dodge bank conflicts
  const int t = threadIdx.x;
  const int n0 = blockIdx.x * 32, k0 = blockIdx.y * 32;
  {
    int kr = t >> 3, nc = t & 7;  // read 4 floats along n, coalesced
    float4 wv = *(const float4*)(W + (size_t)(k0 + kr) * N + n0 + nc * 4);
    Ts[(nc * 4 + 0) * 36 + kr] = f2h(wv.x);
    Ts[(nc * 4 + 1) * 36 + kr] = f2h(wv.y);
    Ts[(nc * 4 + 2) * 36 + kr] = f2h(wv.z);
    Ts[(nc * 4 + 3) * 36 + kr] = f2h(wv.w);
  }
  __syncthreads();
  {
    int nr = t >> 3, kc = t & 7;  // write 4 halves along k, coalesced
    union { unsigned short us[4]; uint2 u; } pk;
    pk.us[0] = Ts[nr * 36 + kc * 4 + 0];
    pk.us[1] = Ts[nr * 36 + kc * 4 + 1];
    pk.us[2] = Ts[nr * 36 + kc * 4 + 2];
    pk.us[3] = Ts[nr * 36 + kc * 4 + 3];
    *(uint2*)(Wt + (size_t)(n0 + nr) * 1024 + k0 + kc * 4) = pk.u;
  }
}

// ---------------------------------------------------------------------------
// Kernel 1: qkv = Xh(f16 [8192,1024]) @ Wt(f16 [3072,1024] pre-transposed)
// Double-buffered async-to-LDS staging. Epilogue scatters Q/K [B,H,T,D] and
// V pre-transposed [B,H,D,T]; Q scaled by 1/sqrt(D).
// ---------------------------------------------------------------------------
__global__ __launch_bounds__(256) void qkv_gemm_kernel(
    const _Float16* __restrict__ Xh, const _Float16* __restrict__ Wt,
    _Float16* __restrict__ Q, _Float16* __restrict__ Kq,
    _Float16* __restrict__ Vt) {
  __shared__ unsigned short As[2][128 * 32];  // [m][k]
  __shared__ unsigned short Bs[2][64 * 32];   // [n][k]
  const int tid = threadIdx.x;
  const int wave = tid >> 5, lane = tid & 31;
  const int bm = blockIdx.y, bn = blockIdx.x;
  const int wm = (wave & 3) * 32, wn = (wave >> 2) * 32;
  const int l16 = lane & 15;
  const int c0 = (lane >> 4) * 8;    // A-operand k-chunk base
  const int h0 = (lane >> 4) * 8;    // B-operand k-chunk base
  const int mloc = (lane >> 4) << 3; // C-layout row offset

  auto issue = [&](int buf, int k0) {
#pragma unroll
    for (int i = 0; i < 2; ++i) {  // A tile: 128x32 halves = 512 x b128
      int idx = tid + i * 256;
      int row = idx >> 2, c8 = idx & 3;
      async_b128(lds_off(&As[buf][row * 32 + c8 * 8]),
                 Xh + (size_t)(bm * 128 + row) * 1024 + k0 + c8 * 8);
    }
    {  // B tile: 64x32 halves = 256 x b128
      int row = tid >> 2, c8 = tid & 3;
      async_b128(lds_off(&Bs[buf][row * 32 + c8 * 8]),
                 Wt + (size_t)(bn * 64 + row) * 1024 + k0 + c8 * 8);
    }
  };

  v8f acc[2][2] = {};
  issue(0, 0);
  for (int k0 = 0; k0 < 1024; k0 += 32) {
    const int cur = (k0 >> 5) & 1;
    wait_async0();
    __syncthreads();
    if (k0 + 32 < 1024) issue(cur ^ 1, k0 + 32);  // overlap copy with compute

    v16h a[2], b[2];
#pragma unroll
    for (int i = 0; i < 2; ++i) {
      int row = wm + i * 16 + l16;
      V16H t;
      t.u[0] = *(const uint4*)&As[cur][row * 32 + c0];
      t.u[1] = *(const uint4*)&As[cur][row * 32 + c0 + 16];
      a[i] = t.v;
    }
#pragma unroll
    for (int j = 0; j < 2; ++j) {
      int row = wn + j * 16 + l16;
      V16H t;
      t.u[0] = *(const uint4*)&Bs[cur][row * 32 + h0];
      t.u[1] = *(const uint4*)&Bs[cur][row * 32 + h0 + 16];
      b[j] = t.v;
    }
#pragma unroll
    for (int i = 0; i < 2; ++i)
#pragma unroll
      for (int j = 0; j < 2; ++j)
        acc[i][j] = wmma_f16(a[i], b[j], acc[i][j]);
  }

#pragma unroll
  for (int i = 0; i < 2; ++i)
#pragma unroll
    for (int j = 0; j < 2; ++j)
#pragma unroll
      for (int r = 0; r < 8; ++r) {
        int m = bm * 128 + wm + i * 16 + r + mloc;
        int n = bn * 64 + wn + j * 16 + l16;
        float val = acc[i][j][r];
        int sel = n >> 10, cc = n & 1023;
        int hh = cc >> 6, d = cc & 63;
        int bb = m >> 11, t = m & 2047;
        size_t bh = (size_t)bb * DIM_H + hh;
        if (sel == 0) {
          Q[(bh * DIM_T + t) * DIM_D + d] = (_Float16)(val * 0.125f);
        } else if (sel == 1) {
          Kq[(bh * DIM_T + t) * DIM_D + d] = (_Float16)val;
        } else {
          Vt[(bh * DIM_D + d) * DIM_T + t] = (_Float16)val;  // pre-transposed
        }
      }
}

// ---------------------------------------------------------------------------
// Kernel 2: fused causal flash-attention per (b,h).
// Q/K f16 [B,H,T,D]; V f16 [B,H,D,T] (pre-transposed).
// K/V tiles double-buffered via async-to-LDS; P via per-wave LDS round-trip.
// ---------------------------------------------------------------------------
__global__ __launch_bounds__(256) void attn_kernel(
    const _Float16* __restrict__ Q, const _Float16* __restrict__ Km,
    const _Float16* __restrict__ Vt, _Float16* __restrict__ O) {
  __shared__ unsigned short Ks[2][32 * 64];   // [key][d]
  __shared__ unsigned short Vs[2][64 * 32];   // [d][key]
  __shared__ unsigned short Ps[8 * 16 * 32];  // per-wave P scratch [m][k]
  const int tid = threadIdx.x, wave = tid >> 5, lane = tid & 31;
  const int l16 = lane & 15;
  const int c0 = (lane >> 4) * 8;
  const int h0 = (lane >> 4) * 8;
  const int mloc = (lane >> 4) << 3;
  const int bh = blockIdx.y;
  const int qb = blockIdx.x * 128;
  const int qw = qb + wave * 16;

  auto issue_kv = [&](int buf, int kb) {
    {  // K block 32x64 halves, natural layout
      int kr = tid >> 3, d8 = tid & 7;
      async_b128(lds_off(&Ks[buf][kr * 64 + d8 * 8]),
                 Km + ((size_t)bh * DIM_T + kb * 32 + kr) * DIM_D + d8 * 8);
    }
    {  // V block 64x32 halves from [B,H,D,T]: pure copy of d-rows
      int dr = tid >> 2, c8 = tid & 3;
      async_b128(lds_off(&Vs[buf][dr * 32 + c8 * 8]),
                 Vt + ((size_t)bh * DIM_D + dr) * DIM_T + kb * 32 + c8 * 8);
    }
  };

  // Q A-operands (two k-chunks over D=64); Q already scaled by 1/sqrt(D).
  v16h qa[2];
  {
    const unsigned short* Qp =
        (const unsigned short*)Q + ((size_t)bh * DIM_T + qw + l16) * DIM_D;
#pragma unroll
    for (int kc = 0; kc < 2; ++kc) {
      V16H t;
      t.u[0] = *(const uint4*)(Qp + kc * 32 + c0);
      t.u[1] = *(const uint4*)(Qp + kc * 32 + c0 + 16);
      qa[kc] = t.v;
    }
  }

  v8f o[4] = {};
  float mrow[8], lrow[8];
#pragma unroll
  for (int r = 0; r < 8; ++r) { mrow[r] = -1e30f; lrow[r] = 0.f; }

  const int kb_end = (qb + 128) >> 5;
  issue_kv(0, 0);
  for (int kb = 0; kb < kb_end; ++kb) {
    const int cur = kb & 1;
    wait_async0();
    __syncthreads();
    if (kb + 1 < kb_end) issue_kv(cur ^ 1, kb + 1);

    if (kb * 32 <= qw + 15) {  // wave-uniform causal skip (no barriers inside)
      v8f s[2] = {};
#pragma unroll
      for (int nt = 0; nt < 2; ++nt)
#pragma unroll
        for (int kc = 0; kc < 2; ++kc) {
          int base = (nt * 16 + l16) * 64 + kc * 32 + h0;
          V16H t;
          t.u[0] = *(const uint4*)&Ks[cur][base];
          t.u[1] = *(const uint4*)&Ks[cur][base + 16];
          s[nt] = wmma_f16(qa[kc], t.v, s[nt]);
        }

      const bool needmask = (kb * 32 + 31) > qw;
      const int n0 = kb * 32 + l16;
#pragma unroll
      for (int r = 0; r < 8; ++r) {
        float s0 = s[0][r], s1 = s[1][r];
        if (needmask) {
          int mg = qw + r + mloc;
          if (n0 > mg) s0 = -1e30f;
          if (n0 + 16 > mg) s1 = -1e30f;
        }
        float mx = fmaxf(s0, s1);
#pragma unroll
        for (int off = 1; off < 16; off <<= 1)
          mx = fmaxf(mx, __shfl_xor(mx, off, 16));
        float mn = fmaxf(mrow[r], mx);
        float alpha = __expf(mrow[r] - mn);
        float p0 = __expf(s0 - mn), p1 = __expf(s1 - mn);
        float rs = p0 + p1;
#pragma unroll
        for (int off = 1; off < 16; off <<= 1) rs += __shfl_xor(rs, off, 16);
        lrow[r] = lrow[r] * alpha + rs;
        mrow[r] = mn;
#pragma unroll
        for (int j = 0; j < 4; ++j) o[j][r] *= alpha;
        // C-layout -> LDS [m][k] for A-layout reload
        unsigned short* pw = &Ps[wave * 512 + (r + mloc) * 32 + l16];
        pw[0]  = f2h(p0);
        pw[16] = f2h(p1);
      }
      // same-wave cross-lane LDS RAW: DS pipe is in-order; fence compiler+hw
      asm volatile("s_wait_dscnt 0" ::: "memory");
      v16h pa;
      {
        V16H t;
        t.u[0] = *(const uint4*)&Ps[wave * 512 + l16 * 32 + c0];
        t.u[1] = *(const uint4*)&Ps[wave * 512 + l16 * 32 + c0 + 16];
        pa = t.v;
      }
#pragma unroll
      for (int j = 0; j < 4; ++j) {  // O += P * V
        int base = (j * 16 + l16) * 32 + h0;
        V16H t;
        t.u[0] = *(const uint4*)&Vs[cur][base];
        t.u[1] = *(const uint4*)&Vs[cur][base + 16];
        o[j] = wmma_f16(pa, t.v, o[j]);
      }
    }
  }

  // Finalize: O /= l, write f16 [B,T,C]
  const int bb = bh >> 4, hh = bh & 15;
#pragma unroll
  for (int r = 0; r < 8; ++r) {
    float inv = 1.0f / lrow[r];
    int t = qw + r + mloc;
#pragma unroll
    for (int j = 0; j < 4; ++j) {
      int n = hh * DIM_D + j * 16 + l16;
      O[((size_t)bb * DIM_T + t) * DIM_C + n] = (_Float16)(o[j][r] * inv);
    }
  }
}

// ---------------------------------------------------------------------------
// Kernel 3: out = attn(f16 [8192,1024]) @ Wot(f16 [1024,1024] pre-transposed)
// ---------------------------------------------------------------------------
__global__ __launch_bounds__(256) void out_gemm_kernel(
    const _Float16* __restrict__ A, const _Float16* __restrict__ Wt,
    float* __restrict__ Out) {
  __shared__ unsigned short As[2][128 * 32];
  __shared__ unsigned short Bs[2][64 * 32];
  const int tid = threadIdx.x;
  const int wave = tid >> 5, lane = tid & 31;
  const int bm = blockIdx.y, bn = blockIdx.x;
  const int wm = (wave & 3) * 32, wn = (wave >> 2) * 32;
  const int l16 = lane & 15;
  const int c0 = (lane >> 4) * 8;
  const int h0 = (lane >> 4) * 8;
  const int mloc = (lane >> 4) << 3;

  auto issue = [&](int buf, int k0) {
#pragma unroll
    for (int i = 0; i < 2; ++i) {
      int idx = tid + i * 256;
      int row = idx >> 2, c8 = idx & 3;
      async_b128(lds_off(&As[buf][row * 32 + c8 * 8]),
                 A + (size_t)(bm * 128 + row) * 1024 + k0 + c8 * 8);
    }
    {
      int row = tid >> 2, c8 = tid & 3;
      async_b128(lds_off(&Bs[buf][row * 32 + c8 * 8]),
                 Wt + (size_t)(bn * 64 + row) * 1024 + k0 + c8 * 8);
    }
  };

  v8f acc[2][2] = {};
  issue(0, 0);
  for (int k0 = 0; k0 < 1024; k0 += 32) {
    const int cur = (k0 >> 5) & 1;
    wait_async0();
    __syncthreads();
    if (k0 + 32 < 1024) issue(cur ^ 1, k0 + 32);

    v16h a[2], b[2];
#pragma unroll
    for (int i = 0; i < 2; ++i) {
      int row = wm + i * 16 + l16;
      V16H t;
      t.u[0] = *(const uint4*)&As[cur][row * 32 + c0];
      t.u[1] = *(const uint4*)&As[cur][row * 32 + c0 + 16];
      a[i] = t.v;
    }
#pragma unroll
    for (int j = 0; j < 2; ++j) {
      int row = wn + j * 16 + l16;
      V16H t;
      t.u[0] = *(const uint4*)&Bs[cur][row * 32 + h0];
      t.u[1] = *(const uint4*)&Bs[cur][row * 32 + h0 + 16];
      b[j] = t.v;
    }
#pragma unroll
    for (int i = 0; i < 2; ++i)
#pragma unroll
      for (int j = 0; j < 2; ++j)
        acc[i][j] = wmma_f16(a[i], b[j], acc[i][j]);
  }

#pragma unroll
  for (int i = 0; i < 2; ++i)
#pragma unroll
    for (int j = 0; j < 2; ++j)
#pragma unroll
      for (int r = 0; r < 8; ++r) {
        int m = bm * 128 + wm + i * 16 + r + mloc;
        int n = bn * 64 + wn + j * 16 + l16;
        Out[(size_t)m * 1024 + n] = acc[i][j][r];
      }
}

// ---------------------------------------------------------------------------
extern "C" void kernel_launch(void* const* d_in, const int* in_sizes, int n_in,
                              void* d_out, int out_size, void* d_ws, size_t ws_size,
                              hipStream_t stream) {
  (void)in_sizes; (void)n_in; (void)out_size; (void)ws_size;
  const float* x     = (const float*)d_in[0];   // [4,2048,1024] f32
  const float* w_qkv = (const float*)d_in[1];   // [1024,3072]  f32
  const float* w_out = (const float*)d_in[2];   // [1024,1024]  f32
  float* out = (float*)d_out;                   // [4,2048,1024] f32

  const size_t nBTC = (size_t)DIM_B * DIM_T * DIM_C;  // 8M
  _Float16* Xh   = (_Float16*)d_ws;                   // [8192,1024]
  _Float16* Wq_t = Xh + nBTC;                         // [3072,1024]
  _Float16* Wo_t = Wq_t + (size_t)3 * DIM_C * DIM_C;  // [1024,1024]
  _Float16* Q    = Wo_t + (size_t)DIM_C * DIM_C;      // [B,H,T,D]
  _Float16* Kh   = Q + nBTC;                          // [B,H,T,D]
  _Float16* Vt   = Kh + nBTC;                         // [B,H,D,T]
  _Float16* Aout = Vt + nBTC;                         // [B,T,C]

  cvt_f16_kernel<<<dim3(nBTC / 1024), 256, 0, stream>>>(x, Xh);
  wtrans_kernel<<<dim3(96, 32), 256, 0, stream>>>(w_qkv, Wq_t, 3072);
  wtrans_kernel<<<dim3(32, 32), 256, 0, stream>>>(w_out, Wo_t, 1024);
  // M=8192, N=3072 -> grid(48, 64)
  qkv_gemm_kernel<<<dim3(48, 64), 256, 0, stream>>>(Xh, Wq_t, Q, Kh, Vt);
  // grid(T/128=16, B*H=64)
  attn_kernel<<<dim3(16, 64), 256, 0, stream>>>(Q, Kh, Vt, Aout);
  // M=8192, N=1024 -> grid(16, 64)
  out_gemm_kernel<<<dim3(16, 64), 256, 0, stream>>>(Aout, Wo_t, out);
}
